// VisualPromptEncoder_6408091206131
// MI455X (gfx1250) — matrix operand order repaired and checked by reading, not compiled, not run on hardware
//
#include <hip/hip_runtime.h>
#include <hip/hip_bf16.h>

#define EMBED 4096
#define NTOK  8192      // 8 * 1024
#define PDIM  64

typedef __attribute__((ext_vector_type(16))) __bf16 v16bf;
typedef __attribute__((ext_vector_type(8)))  __bf16 v8bf;
typedef __attribute__((ext_vector_type(8)))  float  v8f;

// async global->LDS copy of 16 bytes (8 bf16), tracked by ASYNCcnt
__device__ __forceinline__ void async_copy_b128(unsigned lds_off, const void* gptr) {
  asm volatile("global_load_async_to_lds_b128 %0, %1, off"
               :: "v"(lds_off), "v"(gptr) : "memory");
}
__device__ __forceinline__ void wait_async0() {
  asm volatile("s_wait_asynccnt 0x0" ::: "memory");
}

// ---------------------------------------------------------------------------
// 0) zero the compaction counter (ws not re-poisoned between replays)
// ---------------------------------------------------------------------------
__global__ void zero_cnt_kernel(int* cnt) { *cnt = 0; }

// ---------------------------------------------------------------------------
// 1) compact indices of type-2 (polygon) tokens
// ---------------------------------------------------------------------------
__global__ __launch_bounds__(256) void compact_kernel(
    const int* __restrict__ types, int* __restrict__ cnt, int* __restrict__ list) {
  int t = blockIdx.x * 256 + threadIdx.x;
  if (types[t] == 2) {
    int p = atomicAdd(cnt, 1);
    list[p] = t;
  }
}

// ---------------------------------------------------------------------------
// 2) convert poly_w2 (fp32) -> bf16 workspace copy, once per call.
//    Removes all cvt work + half the B bytes from the compute-bound GEMM.
// ---------------------------------------------------------------------------
__global__ __launch_bounds__(256) void w2_bf16_kernel(
    const float* __restrict__ w2, __bf16* __restrict__ w2b) {
  int gid = blockIdx.x * 256 + threadIdx.x;     // 8 floats per thread
  const float4* src = (const float4*)w2 + (size_t)gid * 2;
  float4 a = src[0], b = src[1];
  v8bf o;
  o[0] = (__bf16)a.x; o[1] = (__bf16)a.y; o[2] = (__bf16)a.z; o[3] = (__bf16)a.w;
  o[4] = (__bf16)b.x; o[5] = (__bf16)b.y; o[6] = (__bf16)b.z; o[7] = (__bf16)b.w;
  *(v8bf*)(w2b + (size_t)gid * 8) = o;
}

// ---------------------------------------------------------------------------
// 3) point / box experts: tiny dot products, pure bandwidth. 4 outputs/thread.
// ---------------------------------------------------------------------------
__global__ __launch_bounds__(256) void point_box_kernel(
    const float* __restrict__ prompts, const int* __restrict__ types,
    const float* __restrict__ pw, const float* __restrict__ pb,
    const float* __restrict__ bw, const float* __restrict__ bb,
    const float* __restrict__ temb, float* __restrict__ out) {
  int gid   = blockIdx.x * 256 + threadIdx.x;
  int token = gid >> 10;
  int eg    = (gid & 1023) << 2;
  int t     = types[token];
  if (t == 2) return;               // polygon rows written by the WMMA GEMM
  const float* pr = prompts + (size_t)token * PDIM;
  float4 r;
  float* rv = (float*)&r;
  if (t == 0) {
    float p0 = pr[0], p1 = pr[1];
#pragma unroll
    for (int q = 0; q < 4; ++q) {
      int e = eg + q;
      rv[q] = fmaf(p0, pw[e * 2 + 0], fmaf(p1, pw[e * 2 + 1], pb[e] + temb[e]));
    }
  } else {
    float p0 = pr[0], p1 = pr[1], p2 = pr[2], p3 = pr[3];
#pragma unroll
    for (int q = 0; q < 4; ++q) {
      int e = eg + q;
      float acc = bb[e] + temb[EMBED + e];
      acc = fmaf(p0, bw[e * 4 + 0], acc);
      acc = fmaf(p1, bw[e * 4 + 1], acc);
      acc = fmaf(p2, bw[e * 4 + 2], acc);
      acc = fmaf(p3, bw[e * 4 + 3], acc);
      rv[q] = acc;
    }
  }
  *(float4*)(out + (size_t)token * EMBED + eg) = r;
}

// ---------------------------------------------------------------------------
// 4) polygon stage 1: h = x @ w1^T + b1 ; LayerNorm ; ReLU ; cast bf16 -> ws
// ---------------------------------------------------------------------------
__global__ __launch_bounds__(256) void poly_h_kernel(
    const float* __restrict__ prompts, const int* __restrict__ list,
    const int* __restrict__ cnt, const float* __restrict__ w1,
    const float* __restrict__ b1, const float* __restrict__ gamma,
    const float* __restrict__ beta, __bf16* __restrict__ hws) {
  int bi = blockIdx.x;
  if (bi >= *cnt) return;
  int token = list[bi];

  __shared__ float p[PDIM];
  __shared__ float red[256];
  int tid = threadIdx.x;
  if (tid < 16) {
    float4 v = *(const float4*)(prompts + (size_t)token * PDIM + tid * 4);
    p[tid * 4 + 0] = v.x; p[tid * 4 + 1] = v.y;
    p[tid * 4 + 2] = v.z; p[tid * 4 + 3] = v.w;
  }
  __syncthreads();

  float h[16];
  float sum = 0.f, ssq = 0.f;
#pragma unroll
  for (int j = 0; j < 16; ++j) {
    int e = j * 256 + tid;
    const float4* wr = (const float4*)(w1 + (size_t)e * PDIM);
    float acc = b1[e];
#pragma unroll
    for (int k = 0; k < 16; ++k) {
      float4 wv = wr[k];
      acc = fmaf(wv.x, p[4 * k + 0], acc);
      acc = fmaf(wv.y, p[4 * k + 1], acc);
      acc = fmaf(wv.z, p[4 * k + 2], acc);
      acc = fmaf(wv.w, p[4 * k + 3], acc);
    }
    h[j] = acc;
    sum += acc;
    ssq = fmaf(acc, acc, ssq);
  }

  red[tid] = sum; __syncthreads();
  for (int s = 128; s > 0; s >>= 1) { if (tid < s) red[tid] += red[tid + s]; __syncthreads(); }
  float mu = red[0] * (1.0f / EMBED);
  __syncthreads();
  red[tid] = ssq; __syncthreads();
  for (int s = 128; s > 0; s >>= 1) { if (tid < s) red[tid] += red[tid + s]; __syncthreads(); }
  float var = red[0] * (1.0f / EMBED) - mu * mu;
  float rstd = rsqrtf(var + 1e-5f);

#pragma unroll
  for (int j = 0; j < 16; ++j) {
    int e = j * 256 + tid;
    float y = (h[j] - mu) * rstd * gamma[e] + beta[e];
    y = fmaxf(y, 0.f);
    hws[(size_t)bi * EMBED + e] = (__bf16)y;
  }
}

// ---------------------------------------------------------------------------
// 5) polygon stage 2: out[token,e] = h @ w2^T + b2 + temb[2]
//    bf16 WMMA 16x16x32, f32 accum. Block tile 128(M) x 256(N), 8 waves,
//    each wave owns a 64x64 sub-tile (4x4 WMMA accumulators).
//    Double-buffered LDS fed by async global->LDS DMA (ASYNCcnt).
// ---------------------------------------------------------------------------
__global__ __launch_bounds__(256) void poly_gemm2_wmma(
    const __bf16* __restrict__ hws, const __bf16* __restrict__ w2b,
    const float* __restrict__ b2, const float* __restrict__ temb,
    const int* __restrict__ list, const int* __restrict__ cnt,
    float* __restrict__ out) {
  __shared__ __bf16 ldsA[2][128 * 32];   // [buf][m][k]  8 KB each
  __shared__ __bf16 ldsB[2][256 * 32];   // [buf][n][k] 16 KB each

  int count = *cnt;
  int m0 = blockIdx.y * 128;
  if (m0 >= count) return;               // worst-case grid, early exit
  int n0 = blockIdx.x * 256;

  int tid  = threadIdx.x;
  int lane = tid & 31;
  int wave = tid >> 5;
  int wm = (wave & 1) * 64;              // wave M offset in block tile
  int wn = (wave >> 1) * 64;             // wave N offset in block tile

  v8f acc[4][4];
#pragma unroll
  for (int i = 0; i < 4; ++i)
#pragma unroll
    for (int j = 0; j < 4; ++j)
#pragma unroll
      for (int r = 0; r < 8; ++r) acc[i][j][r] = 0.f;

  // staging: A 128 rows x 32k, 16 bf16 per thread; B 256 rows x 32k, one row/thread
  int ar = tid >> 1, ac = (tid & 1) * 16;
  const __bf16* aSrc = hws + (size_t)(m0 + ar) * EMBED + ac;
  const __bf16* bSrc = w2b + (size_t)(n0 + tid) * EMBED;
  unsigned aOff0 = (unsigned)(size_t)&ldsA[0][ar * 32 + ac];
  unsigned aOff1 = (unsigned)(size_t)&ldsA[1][ar * 32 + ac];
  unsigned bOff0 = (unsigned)(size_t)&ldsB[0][tid * 32];
  unsigned bOff1 = (unsigned)(size_t)&ldsB[1][tid * 32];

#define ISSUE_TILE(kt, aO, bO)                                            \
  do {                                                                    \
    const __bf16* ga = aSrc + (kt) * 32;                                  \
    async_copy_b128((aO), ga);                                            \
    async_copy_b128((aO) + 16, ga + 8);                                   \
    const __bf16* gb = bSrc + (kt) * 32;                                  \
    async_copy_b128((bO), gb);                                            \
    async_copy_b128((bO) + 16, gb + 8);                                   \
    async_copy_b128((bO) + 32, gb + 16);                                  \
    async_copy_b128((bO) + 48, gb + 24);                                  \
  } while (0)

  // per-lane fragment addressing (contiguous k order, same for A and B)
  int am = lane & 15;                    // A: lane -> M row
  int ak = (lane < 16) ? 0 : 16;         // A: contiguous K {ak..ak+15}
  int bn = lane & 15;                    // B: lane -> N column
  int bk = (lane < 16) ? 0 : 16;         // B: contiguous K {bk..bk+15}

  const int KT = EMBED / 32;
  ISSUE_TILE(0, aOff0, bOff0);           // prologue: tile 0 -> buffer 0

  for (int kt = 0; kt < KT; ++kt) {
    wait_async0();                       // this wave's tile-kt DMA complete
    __syncthreads();                     // tile kt visible; all waves past kt-1 compute
    if (kt + 1 < KT) {                   // overlap next tile DMA with compute
      if ((kt + 1) & 1) ISSUE_TILE(kt + 1, aOff1, bOff1);
      else              ISSUE_TILE(kt + 1, aOff0, bOff0);
    }
    const __bf16* bufA = ldsA[kt & 1];
    const __bf16* bufB = ldsB[kt & 1];

    v16bf afrag[4];
#pragma unroll
    for (int i = 0; i < 4; ++i) {
      const __bf16* pa = bufA + (wm + i * 16 + am) * 32 + ak;
      v8bf alo = *(const v8bf*)pa;
      v8bf ahi = *(const v8bf*)(pa + 8);
      afrag[i] = __builtin_shufflevector(alo, ahi,
                   0, 1, 2, 3, 4, 5, 6, 7, 8, 9, 10, 11, 12, 13, 14, 15);
    }
#pragma unroll
    for (int j = 0; j < 4; ++j) {
      v16bf b = *(const v16bf*)(bufB + (wn + j * 16 + bn) * 32 + bk);
#pragma unroll
      for (int i = 0; i < 4; ++i) {
        acc[i][j] = __builtin_amdgcn_wmma_f32_16x16x32_bf16(
            false, afrag[i], false, b, (short)0, acc[i][j], false, false);
      }
    }
  }
#undef ISSUE_TILE

  // epilogue: C/D f32 layout -> VGPR r: M = r + 8*(lane>=16), N = lane&15
  int nlane = lane & 15;
  int mhalf = (lane >> 4) * 8;
#pragma unroll
  for (int j = 0; j < 4; ++j) {
    int e0 = n0 + wn + j * 16 + nlane;
    float s = b2[e0] + temb[2 * EMBED + e0];
#pragma unroll
    for (int i = 0; i < 4; ++i) {
#pragma unroll
      for (int r = 0; r < 8; ++r) {
        int mrow = m0 + wm + i * 16 + mhalf + r;
        if (mrow < count) {
          int token = list[mrow];
          out[(size_t)token * EMBED + e0] = acc[i][j][r] + s;
        }
      }
    }
  }
}

// ---------------------------------------------------------------------------
extern "C" void kernel_launch(void* const* d_in, const int* in_sizes, int n_in,
                              void* d_out, int out_size, void* d_ws, size_t ws_size,
                              hipStream_t stream) {
  const float* prompts = (const float*)d_in[0];
  const int*   ptypes  = (const int*)d_in[1];
  const float* point_w = (const float*)d_in[2];
  const float* point_b = (const float*)d_in[3];
  const float* box_w   = (const float*)d_in[4];
  const float* box_b   = (const float*)d_in[5];
  const float* poly_w1 = (const float*)d_in[6];
  const float* poly_b1 = (const float*)d_in[7];
  const float* ln_g    = (const float*)d_in[8];
  const float* ln_b    = (const float*)d_in[9];
  const float* poly_w2 = (const float*)d_in[10];
  const float* poly_b2 = (const float*)d_in[11];
  const float* temb    = (const float*)d_in[12];
  float* out = (float*)d_out;

  // workspace layout
  int*    cnt  = (int*)d_ws;
  int*    list = (int*)((char*)d_ws + 64);
  __bf16* hws  = (__bf16*)((char*)d_ws + 65536);                       // 64 MB
  __bf16* w2b  = (__bf16*)((char*)d_ws + 65536 + (size_t)NTOK * EMBED * 2); // 32 MB

  zero_cnt_kernel<<<1, 1, 0, stream>>>(cnt);
  compact_kernel<<<NTOK / 256, 256, 0, stream>>>(ptypes, cnt, list);
  w2_bf16_kernel<<<(EMBED * (EMBED / 8)) / 256, 256, 0, stream>>>(poly_w2, w2b);
  point_box_kernel<<<(NTOK * (EMBED / 4)) / 256, 256, 0, stream>>>(
      prompts, ptypes, point_w, point_b, box_w, box_b, temb, out);
  poly_h_kernel<<<NTOK, 256, 0, stream>>>(
      prompts, list, cnt, poly_w1, poly_b1, ln_g, ln_b, hws);
  dim3 g2(EMBED / 256, NTOK / 128);                  // worst-case M, early exit
  poly_gemm2_wmma<<<g2, 256, 0, stream>>>(
      hws, w2b, poly_b2, temb, list, cnt, out);
}
